// Retriever_29807073034676
// MI455X (gfx1250) — compile-verified, matrix operand order checked by reading, not run on hardware
//
#include <hip/hip_runtime.h>
#include <stdint.h>

typedef __bf16 bf16_t;
typedef bf16_t v16bf __attribute__((ext_vector_type(16)));
typedef bf16_t v8bf  __attribute__((ext_vector_type(8)));
typedef float  v8f   __attribute__((ext_vector_type(8)));

#define EMB     256
#define KDIM    1044            // PRED_IN
#define KP      1056            // padded to 33*32
#define NSAMP   5
#define NB      (NSAMP*EMB)     // 1280 stacked output columns
#define MT      64              // edges per workgroup
#define THREADS 512             // 16 waves
#define KSTEPS  (KP/32)         // 33
#define NTILES  20              // 320 cols per wave / 16

// dynamic LDS layout (bytes)
#define A_BYTES   (MT * KP * 2)              // 135168
#define BP_BYTES  (NB * 32 * 2)              // 81920 per K-panel
#define OFF_A     0
#define OFF_BP0   (A_BYTES)
#define OFF_BP1   (A_BYTES + BP_BYTES)
#define OFF_RSUM  (A_BYTES + 2 * BP_BYTES)
#define SMEM_BYTES (OFF_RSUM + MT * 4)       // ~299 KB (< 320 KB WGP LDS)

// ---------- helpers ----------
__device__ __forceinline__ bf16_t f2bf(float f) {
  union { float f; uint32_t u; } v; v.f = f;
  uint32_t r = v.u + 0x7FFFu + ((v.u >> 16) & 1u);   // round-to-nearest-even
  unsigned short h = (unsigned short)(r >> 16);
  bf16_t o; __builtin_memcpy(&o, &h, 2); return o;
}

__device__ __forceinline__ uint32_t hash3(uint32_t a, uint32_t b, uint32_t c) {
  uint32_t h = a * 0x9E3779B1u ^ b * 0x85EBCA77u ^ c * 0xC2B2AE3Du;
  h ^= h >> 16; h *= 0x7FEB352Du; h ^= h >> 15; h *= 0x846CA68Bu; h ^= h >> 16;
  return h;
}
__device__ __forceinline__ float gnoise(uint32_t a, uint32_t b, uint32_t c) {
  uint32_t u1 = hash3(a, b, c);
  uint32_t u2 = hash3(a ^ 0x68bc21ebu, b, c);
  float f1 = ((float)u1 + 1.0f) * (1.0f / 4294967296.0f);
  float f2 = ((float)u2) * (1.0f / 4294967296.0f);
  return sqrtf(-2.0f * logf(f1)) * cosf(6.28318530718f * f2);
}
__device__ __forceinline__ float softplusf(float x) {
  return (x > 20.0f) ? x : log1pf(expf(x));
}

// ---------- tiny utility kernels ----------
__global__ void k_zero(float* __restrict__ p, int n) {
  int i = blockIdx.x * blockDim.x + threadIdx.x;
  if (i < n) p[i] = 0.0f;
}

__global__ void k_count(const int* __restrict__ h, const int* __restrict__ t,
                        float* __restrict__ cf, float* __restrict__ cr, int E) {
  int e = blockIdx.x * blockDim.x + threadIdx.x;
  if (e < E) {
    atomicAdd(&cf[t[e]], 1.0f);
    atomicAdd(&cr[h[e]], 1.0f);
  }
}

__global__ void k_scatter(const int* __restrict__ src, const int* __restrict__ dst,
                          const float* __restrict__ x, int xs,
                          float* __restrict__ acc, int E) {
  int e = blockIdx.x * blockDim.x + threadIdx.x;
  if (e >= E) return;
  int s = src[e], d = dst[e];
  atomicAdd(&acc[(size_t)d * 2 + 0], x[(size_t)s * xs + 0]);
  atomicAdd(&acc[(size_t)d * 2 + 1], x[(size_t)s * xs + 1]);
}

__global__ void k_norm(const float* __restrict__ acc, const float* __restrict__ cnt,
                       float* __restrict__ out, int os, int n) {
  int i = blockIdx.x * blockDim.x + threadIdx.x;
  if (i >= n) return;
  float c = fmaxf(cnt[i], 1.0f);
  out[(size_t)i * os + 0] = acc[(size_t)i * 2 + 0] / c;
  out[(size_t)i * os + 1] = acc[(size_t)i * 2 + 1] / c;
}

// ---------- Bayesian weight sampling ----------
__global__ void k_gen_w1(const float* __restrict__ w1mu, const float* __restrict__ w1rho,
                         bf16_t* __restrict__ Bw) {
  int idx = blockIdx.x * blockDim.x + threadIdx.x;
  const int total = NSAMP * EMB * KP;
  if (idx >= total) return;
  int s   = idx / (EMB * KP);
  int rem = idx - s * (EMB * KP);
  int j   = rem / KP;
  int k   = rem - j * KP;
  float v = 0.0f;
  if (k < KDIM) {
    size_t o = (size_t)j * KDIM + k;
    v = w1mu[o] + gnoise((uint32_t)s, (uint32_t)j, (uint32_t)k) * softplusf(w1rho[o]);
  }
  Bw[(size_t)(s * EMB + j) * KP + k] = f2bf(v);
}

__global__ void k_gen_small(const float* __restrict__ b1mu, const float* __restrict__ b1rho,
                            const float* __restrict__ w2mu, const float* __restrict__ w2rho,
                            const float* __restrict__ b2mu, const float* __restrict__ b2rho,
                            float* __restrict__ b1g, float* __restrict__ w2g,
                            float* __restrict__ b2mean) {
  int n = blockIdx.x * blockDim.x + threadIdx.x;
  if (n >= NB) return;
  int s = n / EMB, j = n - s * EMB;
  b1g[n] = b1mu[j] + gnoise(1000u + s, (uint32_t)j, 7u) * softplusf(b1rho[j]);
  w2g[n] = (w2mu[j] + gnoise(2000u + s, (uint32_t)j, 9u) * softplusf(w2rho[j])) * (1.0f / NSAMP);
  if (n == 0) {
    float a = 0.0f;
    for (int ss = 0; ss < NSAMP; ++ss)
      a += b2mu[0] + gnoise(3000u + ss, 0u, 11u) * softplusf(b2rho[0]);
    b2mean[0] = a * (1.0f / NSAMP);
  }
}

// ---------- fused gather + async-staged bf16 WMMA GEMM + layer-2 epilogue ----------
__device__ __forceinline__ void stage_panel_async(const bf16_t* __restrict__ Bw, int kk,
                                                  uint32_t lds_panel_off, int tid) {
  // copy B K-panel [NB cols][32 K] -> LDS, 16B chunks, 5120 chunks / 512 thr = 10 each
  #pragma unroll
  for (int j = 0; j < 10; ++j) {
    int i  = tid + j * THREADS;
    int col = i >> 2;
    int ch  = i & 3;
    uint64_t g = (uint64_t)(uintptr_t)(Bw + (size_t)col * KP + kk) + (uint32_t)(ch * 16);
    uint32_t l = lds_panel_off + (uint32_t)col * 64u + (uint32_t)(ch * 16);
    asm volatile("global_load_async_to_lds_b128 %0, %1, off"
                 :: "v"(l), "v"(g) : "memory");
  }
}

__global__ __launch_bounds__(THREADS)
void k_fused_gemm(const int* __restrict__ hid, const int* __restrict__ rid,
                  const int* __restrict__ tids,
                  const float* __restrict__ q, const float* __restrict__ ent,
                  const float* __restrict__ nontext, const float* __restrict__ rel,
                  const float* __restrict__ topic, const float* __restrict__ dde,
                  const bf16_t* __restrict__ Bw,
                  const float* __restrict__ b1g, const float* __restrict__ w2g,
                  const float* __restrict__ b2mean,
                  float* __restrict__ out, int E, int n_text) {
  extern __shared__ __align__(16) char smem[];
  bf16_t* At     = (bf16_t*)(smem + OFF_A);      // [MT][KP]
  float*  rowsum = (float*)(smem + OFF_RSUM);    // [MT]
  const uint32_t lds_base = (uint32_t)__builtin_amdgcn_groupstaticsize();

  const int tid  = threadIdx.x;
  const int lane = tid & 31;
  const int wid  = tid >> 5;                     // 0..15
  const int base = blockIdx.x * MT;

  if (tid < MT) rowsum[tid] = 0.0f;

  // ---- build A tile: 8 threads per edge-row, 132 contiguous cols each ----
  {
    const int row = tid >> 3;                    // 0..63
    const int sub = tid & 7;
    int e = base + row; if (e >= E) e = E - 1;
    const int h = hid[e], r = rid[e], t = tids[e];
    const float* enth = (h < n_text) ? (ent + (size_t)h * EMB) : nontext;
    const float* entt = (t < n_text) ? (ent + (size_t)t * EMB) : nontext;
    const float* relr = rel + (size_t)r * EMB;
    bf16_t* arow_p = At + (size_t)row * KP;
    const int c0 = sub * (KP / 8);
    #pragma unroll 4
    for (int c = c0; c < c0 + (KP / 8); ++c) {
      float v;
      if      (c < 256)  v = q[c];                                // h_q
      else if (c < 512)  v = enth[c - 256];                       // h_e[h]: embedding
      else if (c < 514)  v = topic[(size_t)h * 2 + (c - 512)];    // h_e[h]: topic
      else if (c < 522)  v = dde[(size_t)h * 8 + (c - 514)];      // h_e[h]: DDE
      else if (c < 778)  v = relr[c - 522];                       // h_r
      else if (c < 1034) v = entt[c - 778];                       // h_e[t]: embedding
      else if (c < 1036) v = topic[(size_t)t * 2 + (c - 1034)];   // h_e[t]: topic
      else if (c < 1044) v = dde[(size_t)t * 8 + (c - 1036)];     // h_e[t]: DDE
      else               v = 0.0f;                                // K pad
      arow_p[c] = f2bf(v);
    }
  }

  // kick off async staging of first B K-panel while A stores drain
  stage_panel_async(Bw, 0, lds_base + OFF_BP0, tid);

  // wave tiling: 4 waves in M (16 rows each), 4 waves in N (320 cols each)
  const int mgroup = wid & 3;
  const int ngroup = wid >> 2;
  const int arow   = mgroup * 16 + (lane & 15);
  const int koff   = (lane >> 4) * 8;            // bf16 A/B fragment K split
  const int nlane  = ngroup * 320 + (lane & 15);

  v8f acc[NTILES];
  #pragma unroll
  for (int t = 0; t < NTILES; ++t) acc[t] = (v8f)(0.0f);

  const bf16_t* const Ap = At + (size_t)arow * KP + koff;

  for (int ks = 0; ks < KSTEPS; ++ks) {
    const int kk = ks * 32;
    // prefetch next panel into the other buffer
    if (ks + 1 < KSTEPS) {
      stage_panel_async(Bw, kk + 32, lds_base + ((ks + 1) & 1 ? OFF_BP1 : OFF_BP0), tid);
      asm volatile("s_wait_asynccnt 0xa" ::: "memory");   // current panel's 10 ops done
    } else {
      asm volatile("s_wait_asynccnt 0x0" ::: "memory");
    }
    __syncthreads();   // publish current panel (and A tile on first pass)

    const bf16_t* Bp = (const bf16_t*)(smem + ((ks & 1) ? OFF_BP1 : OFF_BP0));

    v16bf a;
    {
      v8bf lo = *(const v8bf*)(Ap + kk);         // K {0..7} / {8..15}
      v8bf hi = *(const v8bf*)(Ap + kk + 16);    // K {16..23} / {24..31}
      #pragma unroll
      for (int i = 0; i < 8; ++i) { a[i] = lo[i]; a[i + 8] = hi[i]; }
    }
    #pragma unroll
    for (int t = 0; t < NTILES; ++t) {
      const bf16_t* bp = Bp + (size_t)(nlane + t * 16) * 32 + koff;
      v16bf b;
      v8bf blo = *(const v8bf*)(bp);
      v8bf bhi = *(const v8bf*)(bp + 16);
      #pragma unroll
      for (int i = 0; i < 8; ++i) { b[i] = blo[i]; b[i + 8] = bhi[i]; }
      acc[t] = __builtin_amdgcn_wmma_f32_16x16x32_bf16(
          false, a, false, b, (short)0, acc[t], false, false);
    }
    __syncthreads();   // all reads of this panel done before its buffer is rewritten
  }

  // ---- epilogue: out_row = sum_n relu(acc + b1[n]) * (w2[n]/5) ----
  float partial[8];
  #pragma unroll
  for (int i = 0; i < 8; ++i) partial[i] = 0.0f;
  #pragma unroll
  for (int t = 0; t < NTILES; ++t) {
    const int n = nlane + t * 16;
    const float bias = b1g[n];
    const float w2v  = w2g[n];
    #pragma unroll
    for (int i = 0; i < 8; ++i) {
      float v = acc[t][i] + bias;
      v = v > 0.0f ? v : 0.0f;
      partial[i] += v * w2v;
    }
  }
  #pragma unroll
  for (int i = 0; i < 8; ++i) {
    float p = partial[i];
    p += __shfl_xor(p, 1, 32);
    p += __shfl_xor(p, 2, 32);
    p += __shfl_xor(p, 4, 32);
    p += __shfl_xor(p, 8, 32);
    if ((lane & 15) == 0) {
      const int rrow = mgroup * 16 + ((lane & 16) ? 8 : 0) + i;   // C layout: vgpr->M
      atomicAdd(&rowsum[rrow], p);
    }
  }
  __syncthreads();
  if (tid < MT) {
    const int e = base + tid;
    if (e < E) out[e] = rowsum[tid] + b2mean[0];
  }
}

// ---------- host launcher ----------
extern "C" void kernel_launch(void* const* d_in, const int* in_sizes, int n_in,
                              void* d_out, int out_size, void* d_ws, size_t ws_size,
                              hipStream_t stream) {
  const int* hid   = (const int*)d_in[0];
  const int* rid   = (const int*)d_in[1];
  const int* tids  = (const int*)d_in[2];
  const float* q       = (const float*)d_in[3];
  const float* ent     = (const float*)d_in[4];
  const float* rel     = (const float*)d_in[6];
  const float* topic   = (const float*)d_in[7];
  const float* nontext = (const float*)d_in[8];
  const float* w1mu  = (const float*)d_in[9];
  const float* w1rho = (const float*)d_in[10];
  const float* b1mu  = (const float*)d_in[11];
  const float* b1rho = (const float*)d_in[12];
  const float* w2mu  = (const float*)d_in[13];
  const float* w2rho = (const float*)d_in[14];
  const float* b2mu  = (const float*)d_in[15];
  const float* b2rho = (const float*)d_in[16];
  float* out = (float*)d_out;

  const int E      = in_sizes[0];
  const int n_text = in_sizes[4] / EMB;
  const int N      = in_sizes[7] / 2;

  // workspace layout
  char*  ws  = (char*)d_ws;
  size_t off = 0;
  auto take = [&](size_t bytes) -> void* {
    void* p = ws + off;
    off = (off + bytes + 255) & ~(size_t)255;
    return p;
  };
  bf16_t* Bw     = (bf16_t*)take((size_t)NB * KP * sizeof(bf16_t));
  float*  b1g    = (float*)take(NB * sizeof(float));
  float*  w2g    = (float*)take(NB * sizeof(float));
  float*  b2mean = (float*)take(16);
  float*  cf     = (float*)take((size_t)N * sizeof(float));
  float*  cr     = (float*)take((size_t)N * sizeof(float));
  float*  dde    = (float*)take((size_t)N * 8 * sizeof(float));
  float*  acc    = (float*)take((size_t)N * 2 * sizeof(float));
  (void)ws_size; (void)n_in; (void)out_size;

  const int B256 = 256;
  auto blocks = [](long long n) { return (int)((n + 255) / 256); };

  // allow large dynamic LDS (no-op if already permitted)
  static int attr_done = 0;   // idempotent host-side attribute set (not a stream op)
  if (!attr_done) {
    (void)hipFuncSetAttribute((const void*)k_fused_gemm,
                              hipFuncAttributeMaxDynamicSharedMemorySize, SMEM_BYTES);
    attr_done = 1;
  }

  // 1) sample Bayesian weights (stacked, bf16)
  k_gen_w1<<<blocks((long long)NSAMP * EMB * KP), B256, 0, stream>>>(w1mu, w1rho, Bw);
  k_gen_small<<<blocks(NB), B256, 0, stream>>>(b1mu, b1rho, w2mu, w2rho, b2mu, b2rho,
                                               b1g, w2g, b2mean);

  // 2) degree counts
  k_zero<<<blocks(N), B256, 0, stream>>>(cf, N);
  k_zero<<<blocks(N), B256, 0, stream>>>(cr, N);
  k_count<<<blocks(E), B256, 0, stream>>>(hid, tids, cf, cr, E);

  // 3) DDE rounds: fwd(h->t) x2, rev(t->h) x2 -> dde[N][8]
  k_zero<<<blocks(2LL * N), B256, 0, stream>>>(acc, 2 * N);
  k_scatter<<<blocks(E), B256, 0, stream>>>(hid, tids, topic, 2, acc, E);
  k_norm<<<blocks(N), B256, 0, stream>>>(acc, cf, dde + 0, 8, N);

  k_zero<<<blocks(2LL * N), B256, 0, stream>>>(acc, 2 * N);
  k_scatter<<<blocks(E), B256, 0, stream>>>(hid, tids, dde + 0, 8, acc, E);
  k_norm<<<blocks(N), B256, 0, stream>>>(acc, cf, dde + 2, 8, N);

  k_zero<<<blocks(2LL * N), B256, 0, stream>>>(acc, 2 * N);
  k_scatter<<<blocks(E), B256, 0, stream>>>(tids, hid, topic, 2, acc, E);
  k_norm<<<blocks(N), B256, 0, stream>>>(acc, cr, dde + 4, 8, N);

  k_zero<<<blocks(2LL * N), B256, 0, stream>>>(acc, 2 * N);
  k_scatter<<<blocks(E), B256, 0, stream>>>(tids, hid, dde + 4, 8, acc, E);
  k_norm<<<blocks(N), B256, 0, stream>>>(acc, cr, dde + 6, 8, N);

  // 4) fused gather + async-staged WMMA GEMM + epilogue
  const int ngrid = (E + MT - 1) / MT;
  k_fused_gemm<<<ngrid, THREADS, SMEM_BYTES, stream>>>(
      hid, rid, tids, q, ent, nontext, rel, topic, dde, Bw, b1g, w2g, b2mean,
      out, E, n_text);
}
